// WindowAttention_58342835749597
// MI455X (gfx1250) — matrix operand order repaired and checked by reading, not compiled, not run on hardware
//
#include <hip/hip_runtime.h>
#include <hip/hip_bf16.h>

// ---------------------------------------------------------------------------
// WindowAttention on MI455X (gfx1250): all GEMMs via v_wmma_f32_16x16x32_f16.
//
// Shapes: B=16, H=64, W=64, C=256, HEADS=8, D=32, INNER=256, T=B*H*W=65536.
// ---------------------------------------------------------------------------

#define BATCH   16
#define HEIGHT  64
#define WIDTH   64
#define CMODEL  256
#define NHEADS  8
#define HDIM    32
#define INNER_  256
#define NTOK    (BATCH * HEIGHT * WIDTH)   // 65536

typedef _Float16 v8h  __attribute__((ext_vector_type(8)));
typedef _Float16 v16h __attribute__((ext_vector_type(16)));
typedef float    v8f  __attribute__((ext_vector_type(8)));

// --- WMMA helper: D = A(16x32 f16) x B(32x16 f16) + C(16x16 f32) -----------
__device__ __forceinline__ v8f wmma16x16x32(v16h a, v16h b, v8f c) {
  return __builtin_amdgcn_wmma_f32_16x16x32_f16(
      /*neg_a=*/false, a, /*neg_b=*/false, b,
      /*c_mod=*/(short)0, c, /*reuse_a=*/false, /*reuse_b=*/false);
}

__device__ __forceinline__ v16h cat8(v8h lo, v8h hi) {
  v16h r;
#pragma unroll
  for (int j = 0; j < 8; ++j) { r[j] = lo[j]; r[j + 8] = hi[j]; }
  return r;
}

// A-fragment: row-major f16 [.., rowStride]; lane holds row m, halves of K.
// a[j<8]  = A[m][kk + 8h + j]
// a[j>=8] = A[m][kk + 16 + 8h + (j-8)]
__device__ __forceinline__ v16h load_a_frag(const _Float16* __restrict__ base,
                                            int m, int rowStride, int kk, int hH) {
  const _Float16* p = base + (size_t)m * rowStride + kk + 8 * hH;
  v8h lo = *(const v8h*)p;
  v8h hi = *(const v8h*)(p + 16);
  return cat8(lo, hi);
}

// B-fragment from K-contiguous (i.e. B^T row-major) f16 [N][Kstride]:
// b[j<8] = Bt[n][kk + 8h + j], b[j>=8] = Bt[n][kk + 16 + 8h + (j-8)]
__device__ __forceinline__ v16h load_b_frag(const _Float16* __restrict__ baseT,
                                            int n, int kStride, int kk, int hH) {
  const _Float16* p = baseT + (size_t)n * kStride + kk + 8 * hH;
  v8h lo = *(const v8h*)p;
  v8h hi = *(const v8h*)(p + 16);
  return cat8(lo, hi);
}

// ---------------------------------------------------------------------------
// Kernel 1: convert + transpose weights to f16.
//   wqkvT[mat][out][in] = w{q,k,v}[in][out]   (mat = 0,1,2)
//   woT[out][in]        = wo[in][out]
// ---------------------------------------------------------------------------
__global__ void wa_prep_w(const float* __restrict__ wq, const float* __restrict__ wk,
                          const float* __restrict__ wv, const float* __restrict__ wo,
                          _Float16* __restrict__ wqkvT, _Float16* __restrict__ woT) {
  int i = blockIdx.x * blockDim.x + threadIdx.x;   // [0, 4*65536)
  int mat = i >> 16;
  int rem = i & 65535;
  int nout = rem >> 8;
  int kin  = rem & 255;
  const float* src = (mat == 0) ? wq : (mat == 1) ? wk : (mat == 2) ? wv : wo;
  _Float16 val = (_Float16)src[kin * 256 + nout];
  if (mat < 3) wqkvT[(mat << 16) + (nout << 8) + kin] = val;
  else         woT[(nout << 8) + kin] = val;
}

// ---------------------------------------------------------------------------
// Kernel 2: x fp32 -> f16 (token-major [T][256], identical linear layout).
// ---------------------------------------------------------------------------
__global__ void wa_cvt_x(const float* __restrict__ x, _Float16* __restrict__ xh) {
  int i = blockIdx.x * blockDim.x + threadIdx.x;
  xh[i] = (_Float16)x[i];
}

// ---------------------------------------------------------------------------
// Kernel 3: fused QKV projection.
// One wave computes a 16-token x 64-col strip of [q|k|v] (strip in [0,12)).
// Results scattered to head-major layouts:
//   q,k : [b][head][h][w][d]   (row-major over (w,d))
//   v   : [b][head][h][d][w]   (transposed so it is B^T for P@V)
// ---------------------------------------------------------------------------
__global__ void wa_qkv(const _Float16* __restrict__ xh, const _Float16* __restrict__ wT,
                       _Float16* __restrict__ q, _Float16* __restrict__ k,
                       _Float16* __restrict__ v) {
  int wave = blockIdx.x * (blockDim.x >> 5) + (threadIdx.x >> 5);
  int lane = threadIdx.x & 31;
  int ln = lane & 15, hH = lane >> 4;

  int strip = wave % 12;           // 12 strips of 64 cols over 768 outputs
  int mTile = wave / 12;           // 4096 token tiles
  int rowBase = mTile * 16;

  const _Float16* aBase = xh + (size_t)rowBase * CMODEL;
  const _Float16* bBase = wT + (size_t)strip * 64 * CMODEL;

  v8f acc[4] = {};
#pragma unroll
  for (int kk = 0; kk < CMODEL; kk += 32) {
    if (kk + 32 < CMODEL)
      __builtin_prefetch((const void*)(aBase + (size_t)ln * CMODEL + kk + 32), 0, 0);
    v16h a = load_a_frag(aBase, ln, CMODEL, kk, hH);
#pragma unroll
    for (int t = 0; t < 4; ++t) {
      v16h b = load_b_frag(bBase, t * 16 + ln, CMODEL, kk, hH);
      acc[t] = wmma16x16x32(a, b, acc[t]);
    }
  }

  int mat  = strip >> 2;            // 0=q 1=k 2=v
  int colW = (strip & 3) * 64;      // col base within the 256-wide matrix
#pragma unroll
  for (int r = 0; r < 8; ++r) {
    int tok = rowBase + r + 8 * hH;           // C-frag row M = r + 8h
    int b_  = tok >> 12;                      // / (H*W)
    int hh  = (tok >> 6) & 63;
    int ww  = tok & 63;
    int base_bh = ((b_ * NHEADS) * HEIGHT + hh);   // head added per column below
#pragma unroll
    for (int t = 0; t < 4; ++t) {
      int col  = colW + t * 16 + ln;
      int head = col >> 5;
      int d    = col & 31;
      _Float16 val = (_Float16)acc[t][r];
      size_t rowIdx = (size_t)(base_bh + head * HEIGHT) * (WIDTH * HDIM);
      if (mat == 0)      q[rowIdx + ww * HDIM + d] = val;
      else if (mat == 1) k[rowIdx + ww * HDIM + d] = val;
      else               v[rowIdx + d * WIDTH + ww] = val;
    }
  }
}

// ---------------------------------------------------------------------------
// Kernel 4: attention per (b, head, h) row.  seq = 64, head_dim = 32.
// Block = 128 threads = 4 waves; wave qt handles q-rows [16*qt, 16*qt+16).
// S = q k^T (4 WMMAs, K=32), softmax via shfl_xor within 16-lane halves,
// P staged in LDS (f16), O = P v (2 k-steps x 2 n-tiles WMMAs).
// ---------------------------------------------------------------------------
__global__ void wa_attn(const _Float16* __restrict__ q, const _Float16* __restrict__ k,
                        const _Float16* __restrict__ v, _Float16* __restrict__ inner) {
  __shared__ __attribute__((aligned(16))) _Float16 Plds[4 * 16 * 64];

  int row  = blockIdx.x;               // (b*8 + head)*64 + hh
  int qt   = threadIdx.x >> 5;         // q tile 0..3
  int lane = threadIdx.x & 31;
  int ln = lane & 15, hH = lane >> 4;

  const _Float16* qR = q + (size_t)row * (WIDTH * HDIM);
  const _Float16* kR = k + (size_t)row * (WIDTH * HDIM);
  const _Float16* vR = v + (size_t)row * (HDIM * WIDTH);

  // ---- S = q @ k^T  (K = HDIM = 32, one k-step) ----
  v16h aq = load_a_frag(qR, qt * 16 + ln, HDIM, 0, hH);
  v8f s[4];
#pragma unroll
  for (int t = 0; t < 4; ++t) {
    v16h bk = load_b_frag(kR, t * 16 + ln, HDIM, 0, hH);   // k itself is B^T
    v8f z = {};
    s[t] = wmma16x16x32(aq, bk, z);
  }

  // ---- softmax over the 64 keys (scale folded into exp argument) ----
  const float scale = 0.17677669529663687f;   // 1/sqrt(32)
#pragma unroll
  for (int r = 0; r < 8; ++r) {
    float mx = s[0][r];
#pragma unroll
    for (int t = 1; t < 4; ++t) mx = fmaxf(mx, s[t][r]);
    mx = fmaxf(mx, __shfl_xor(mx, 1, 32));
    mx = fmaxf(mx, __shfl_xor(mx, 2, 32));
    mx = fmaxf(mx, __shfl_xor(mx, 4, 32));
    mx = fmaxf(mx, __shfl_xor(mx, 8, 32));
    float sum = 0.f;
#pragma unroll
    for (int t = 0; t < 4; ++t) {
      float e = __expf((s[t][r] - mx) * scale);
      s[t][r] = e;
      sum += e;
    }
    sum += __shfl_xor(sum, 1, 32);
    sum += __shfl_xor(sum, 2, 32);
    sum += __shfl_xor(sum, 4, 32);
    sum += __shfl_xor(sum, 8, 32);
    float inv = 1.0f / sum;
#pragma unroll
    for (int t = 0; t < 4; ++t) s[t][r] *= inv;
  }

  // ---- stage P to LDS as f16 [16 rows][64 cols] per wave ----
  _Float16* Pw = &Plds[qt * 16 * 64];
#pragma unroll
  for (int t = 0; t < 4; ++t)
#pragma unroll
    for (int r = 0; r < 8; ++r)
      Pw[(r + 8 * hH) * 64 + t * 16 + ln] = (_Float16)s[t][r];
  __syncthreads();

  // ---- O = P @ v : K = 64 (two k-steps), N = 32 (two tiles) ----
  v8f o[2] = {};
  const _Float16* Pp = &Plds[qt * 16 * 64 + ln * 64];
#pragma unroll
  for (int kk = 0; kk < 64; kk += 32) {
    v8h lo = *(const v8h*)(Pp + kk + 8 * hH);
    v8h hi = *(const v8h*)(Pp + kk + 16 + 8 * hH);
    v16h ap = cat8(lo, hi);
#pragma unroll
    for (int t2 = 0; t2 < 2; ++t2) {
      v16h bv = load_b_frag(vR, t2 * 16 + ln, WIDTH, kk, hH);  // vR is v^T
      o[t2] = wmma16x16x32(ap, bv, o[t2]);
    }
  }

  // ---- scatter O back to token-major inner [T][256] ----
  int b_   = row >> 9;                 // / (8*64)
  int head = (row >> 6) & 7;
  int hh   = row & 63;
  int tokenBase = (b_ * HEIGHT + hh) * WIDTH;
#pragma unroll
  for (int r = 0; r < 8; ++r) {
    int tok = tokenBase + qt * 16 + r + 8 * hH;
#pragma unroll
    for (int t2 = 0; t2 < 2; ++t2)
      inner[(size_t)tok * INNER_ + head * HDIM + t2 * 16 + ln] = (_Float16)o[t2][r];
  }
}

// ---------------------------------------------------------------------------
// Kernel 5: output projection + bias -> fp32.
// One wave: 16-token x 64-col strip (strip in [0,4)).
// ---------------------------------------------------------------------------
__global__ void wa_oproj(const _Float16* __restrict__ inner, const _Float16* __restrict__ woT,
                         const float* __restrict__ bo, float* __restrict__ out) {
  int wave = blockIdx.x * (blockDim.x >> 5) + (threadIdx.x >> 5);
  int lane = threadIdx.x & 31;
  int ln = lane & 15, hH = lane >> 4;

  int strip = wave & 3;
  int mTile = wave >> 2;
  int rowBase = mTile * 16;

  const _Float16* aBase = inner + (size_t)rowBase * INNER_;
  const _Float16* bBase = woT + (size_t)strip * 64 * INNER_;

  v8f acc[4] = {};
#pragma unroll
  for (int kk = 0; kk < INNER_; kk += 32) {
    if (kk + 32 < INNER_)
      __builtin_prefetch((const void*)(aBase + (size_t)ln * INNER_ + kk + 32), 0, 0);
    v16h a = load_a_frag(aBase, ln, INNER_, kk, hH);
#pragma unroll
    for (int t = 0; t < 4; ++t) {
      v16h b = load_b_frag(bBase, t * 16 + ln, INNER_, kk, hH);
      acc[t] = wmma16x16x32(a, b, acc[t]);
    }
  }

#pragma unroll
  for (int t = 0; t < 4; ++t) {
    int nout = strip * 64 + t * 16 + ln;
    float bias = bo[nout];
#pragma unroll
    for (int r = 0; r < 8; ++r) {
      int tok = rowBase + r + 8 * hH;
      out[(size_t)tok * CMODEL + nout] = acc[t][r] + bias;
    }
  }
}

// ---------------------------------------------------------------------------
// Launcher.  Inputs: x, wq, wk, wv, wo, bo (all fp32). Output: fp32 [T][256].
// Workspace: xh(32MB) wT(384KB) woT(128KB) q(32MB) k(32MB) v(32MB) inner(32MB)
// ---------------------------------------------------------------------------
extern "C" void kernel_launch(void* const* d_in, const int* in_sizes, int n_in,
                              void* d_out, int out_size, void* d_ws, size_t ws_size,
                              hipStream_t stream) {
  (void)in_sizes; (void)n_in; (void)out_size; (void)ws_size;

  const float* x  = (const float*)d_in[0];
  const float* wq = (const float*)d_in[1];
  const float* wk = (const float*)d_in[2];
  const float* wv = (const float*)d_in[3];
  const float* wo = (const float*)d_in[4];
  const float* bo = (const float*)d_in[5];
  float* out = (float*)d_out;

  char* ws = (char*)d_ws;
  size_t off = 0;
  _Float16* xh    = (_Float16*)(ws + off); off += (size_t)NTOK * CMODEL * 2;     // 32 MB
  _Float16* wT    = (_Float16*)(ws + off); off += (size_t)3 * 256 * 256 * 2;     // 384 KB
  _Float16* woT   = (_Float16*)(ws + off); off += (size_t)256 * 256 * 2;         // 128 KB
  _Float16* qbuf  = (_Float16*)(ws + off); off += (size_t)NTOK * INNER_ * 2;     // 32 MB
  _Float16* kbuf  = (_Float16*)(ws + off); off += (size_t)NTOK * INNER_ * 2;     // 32 MB
  _Float16* vbuf  = (_Float16*)(ws + off); off += (size_t)NTOK * INNER_ * 2;     // 32 MB
  _Float16* inner = (_Float16*)(ws + off); off += (size_t)NTOK * INNER_ * 2;     // 32 MB

  // 1) weights fp32 -> f16 transposed
  wa_prep_w<<<(4 * 65536) / 256, 256, 0, stream>>>(wq, wk, wv, wo, wT, woT);
  // 2) x fp32 -> f16
  wa_cvt_x<<<(NTOK * CMODEL) / 256, 256, 0, stream>>>(x, xh);
  // 3) QKV projection: 4096 m-tiles x 12 strips = 49152 waves, 8 waves/block
  wa_qkv<<<(4096 * 12) / 8, 256, 0, stream>>>(xh, wT, qbuf, kbuf, vbuf);
  // 4) attention: one block per (b, head, h) row
  wa_attn<<<BATCH * NHEADS * HEIGHT, 128, 0, stream>>>(qbuf, kbuf, vbuf, inner);
  // 5) output projection: 4096 m-tiles x 4 strips = 16384 waves, 8 waves/block
  wa_oproj<<<(4096 * 4) / 8, 256, 0, stream>>>(inner, woT, bo, out);
}
